// GraphNeuralCSI_37452114821305
// MI455X (gfx1250) — matrix operand order repaired and checked by reading, not compiled, not run on hardware
//
#include <hip/hip_runtime.h>
#include <hip/hip_bf16.h>

#define H      256
#define LCNT   3
#define NHEADS 4
#define FDIM   64
#define ADIM   8
#define NNODES 512
#define TSTEPS 64
#define BATCH  32
#define NCLS   10
#define EPSV   1e-5f
#define ROWS   (BATCH * NNODES)   // 16384
#define MT     (BATCH * TSTEPS)   // 2048

typedef __bf16 bf16;
typedef bf16  v16bf __attribute__((ext_vector_type(16)));
typedef bf16  v8bf  __attribute__((ext_vector_type(8)));
typedef float v8f   __attribute__((ext_vector_type(8)));

// ---------------------------------------------------------------------------
// CDNA5 async copy: global -> LDS, 16 bytes per lane, tracked by ASYNCcnt.
// Low 32 bits of a generic LDS pointer are the LDS byte address (ISA 10.2).
// ---------------------------------------------------------------------------
__device__ __forceinline__ void async_ld16(const bf16* g, bf16* l)
{
    unsigned lo  = (unsigned)(uintptr_t)l;
    unsigned long long ga = (unsigned long long)(uintptr_t)g;
    asm volatile("global_load_async_to_lds_b128 %0, %1, off"
                 :: "v"(lo), "v"(ga) : "memory");
}
__device__ __forceinline__ void wait_async0()
{
    asm volatile("s_wait_asynccnt 0x0" ::: "memory");
}

// B slab in LDS: 64 columns x 32 K bf16, column stride padded to 40 elems (80B)
#define BCOL 40

// ---------------------------------------------------------------------------
// Generic bf16 GEMM: C[m,n] = sum_k A[m,k]*W[n,k] (+ bias[n])
// A: M x K row-major bf16, W: N x K row-major bf16, C: M x N f32.
// Requires M%128==0, N%64==0, K%32==0. grid = (M/128, N/64), block = 256.
// 8 waves/block, each wave owns a 16x64 tile; the 64x32 weight K-slab is
// staged in LDS once per block via async-to-LDS, double buffered.
// ---------------------------------------------------------------------------
__global__ __launch_bounds__(256) void k_gemm_bf16(
    const bf16* __restrict__ A, const bf16* __restrict__ W,
    const float* __restrict__ bias, float* __restrict__ C,
    int M, int N, int K)
{
    __shared__ bf16 bsm[2][64 * BCOL];

    const int lane  = threadIdx.x & 31;
    const int wave  = threadIdx.x >> 5;
    const int m0    = blockIdx.x * 128 + wave * 16;
    const int n0    = blockIdx.y * 64;
    const int mrow  = m0 + (lane & 15);
    const int kbase = (lane >> 4) * 16;   // A fragment K base
    const int koffB = (lane >> 4) * 16;   // B fragment K base
    const int ncol  = lane & 15;

    // cooperative slab fetch: 256 threads x 16B = 4KB = 64 cols x 64B
    const int lcol = threadIdx.x >> 2;
    const int lkc  = (threadIdx.x & 3) * 8;

    v8f acc0 = {}, acc1 = {}, acc2 = {}, acc3 = {};

    // prologue: stage slab 0
    async_ld16(W + (size_t)(n0 + lcol) * K + lkc, &bsm[0][lcol * BCOL + lkc]);
    wait_async0();
    __syncthreads();

    int buf = 0;
    for (int k0 = 0; k0 < K; k0 += 32) {
        // prefetch next slab into the other buffer (overlaps compute)
        if (k0 + 32 < K)
            async_ld16(W + (size_t)(n0 + lcol) * K + (k0 + 32) + lkc,
                       &bsm[buf ^ 1][lcol * BCOL + lkc]);

        // A fragment (direct from global; each element read exactly once)
        const bf16* ap = A + (size_t)mrow * K + k0;
        v8bf alo = *(const v8bf*)(ap + kbase);
        v8bf ahi = *(const v8bf*)(ap + kbase + 8);
        v16bf af;
#pragma unroll
        for (int e = 0; e < 8; ++e) { af[e] = alo[e]; af[e + 8] = ahi[e]; }

        // B fragments from LDS
        const bf16* bb = &bsm[buf][koffB];
        v16bf b0, b1, b2, b3;
        {
            const bf16* p0 = bb + (0  + ncol) * BCOL;
            const bf16* p1 = bb + (16 + ncol) * BCOL;
            const bf16* p2 = bb + (32 + ncol) * BCOL;
            const bf16* p3 = bb + (48 + ncol) * BCOL;
            v8bf l0 = *(const v8bf*)p0,       h0 = *(const v8bf*)(p0 + 8);
            v8bf l1 = *(const v8bf*)p1,       h1 = *(const v8bf*)(p1 + 8);
            v8bf l2 = *(const v8bf*)p2,       h2 = *(const v8bf*)(p2 + 8);
            v8bf l3 = *(const v8bf*)p3,       h3 = *(const v8bf*)(p3 + 8);
#pragma unroll
            for (int e = 0; e < 8; ++e) {
                b0[e] = l0[e]; b0[e + 8] = h0[e];
                b1[e] = l1[e]; b1[e + 8] = h1[e];
                b2[e] = l2[e]; b2[e + 8] = h2[e];
                b3[e] = l3[e]; b3[e + 8] = h3[e];
            }
        }

        acc0 = __builtin_amdgcn_wmma_f32_16x16x32_bf16(false, af, false, b0, (short)0, acc0, false, false);
        acc1 = __builtin_amdgcn_wmma_f32_16x16x32_bf16(false, af, false, b1, (short)0, acc1, false, false);
        acc2 = __builtin_amdgcn_wmma_f32_16x16x32_bf16(false, af, false, b2, (short)0, acc2, false, false);
        acc3 = __builtin_amdgcn_wmma_f32_16x16x32_bf16(false, af, false, b3, (short)0, acc3, false, false);

        // slab handoff: async copy landed + all waves done with current slab
        wait_async0();
        __syncthreads();
        buf ^= 1;
    }

    const int mhi = (lane >> 4) * 8;
#pragma unroll
    for (int r = 0; r < 8; ++r) {
        const int m = m0 + r + mhi;
        float* crow = C + (size_t)m * N;
        int n;
        n = n0 +  0 + ncol; crow[n] = acc0[r] + (bias ? bias[n] : 0.f);
        n = n0 + 16 + ncol; crow[n] = acc1[r] + (bias ? bias[n] : 0.f);
        n = n0 + 32 + ncol; crow[n] = acc2[r] + (bias ? bias[n] : 0.f);
        n = n0 + 48 + ncol; crow[n] = acc3[r] + (bias ? bias[n] : 0.f);
    }
}

// float -> bf16 conversion for weight matrices
__global__ void k_f2bf(const float* __restrict__ s, bf16* __restrict__ d, int n)
{
    int i = blockIdx.x * blockDim.x + threadIdx.x;
    if (i < n) d[i] = (bf16)s[i];
}

// Input projection for one timestep: h[b,i,c] = x[b,t,f,a]*w_in[c,0] + b_in[c]
__global__ void k_in_proj(const float* __restrict__ x, const float* __restrict__ w_in,
                          const float* __restrict__ b_in, bf16* __restrict__ hbf, int t)
{
    int r = blockIdx.x;           // global row = b*NNODES + i
    int c = threadIdx.x;
    int b = r >> 9, i = r & (NNODES - 1);
    int f = i >> 3, a = i & (ADIM - 1);
    float xv = x[(((size_t)b * TSTEPS + t) * FDIM + f) * ADIM + a];
    float h  = xv * w_in[c * 2 + 0] + b_in[c];
    hbf[(size_t)r * H + c] = (bf16)h;
}

__global__ void k_zero(float* __restrict__ p, int n)
{
    int i = blockIdx.x * blockDim.x + threadIdx.x;
    if (i < n) p[i] = 0.f;
}

// Sparse adjacency aggregation (<=9 neighbors) + bias; accumulate BN partials.
__global__ __launch_bounds__(256) void k_aggregate(
    const float* __restrict__ s, const float* __restrict__ adj,
    const float* __restrict__ bias, float* __restrict__ o, float* __restrict__ stats)
{
    int c  = threadIdx.x;
    int r0 = blockIdx.x * 8;
    float sum = 0.f, ssq = 0.f;
    for (int rr = 0; rr < 8; ++rr) {
        int r = r0 + rr;
        int b = r >> 9, i = r & (NNODES - 1);
        int f = i >> 3, a = i & (ADIM - 1);
        float acc = 0.f;
#pragma unroll
        for (int df = -1; df <= 1; ++df) {
            int fn = f + df;
            if ((unsigned)fn >= FDIM) continue;
#pragma unroll
            for (int da = -1; da <= 1; ++da) {
                int an = a + da;
                if ((unsigned)an >= ADIM) continue;
                int j = fn * ADIM + an;
                float w = adj[(size_t)i * NNODES + j];   // broadcast load
                acc += w * s[((size_t)b * NNODES + j) * H + c];
            }
        }
        float v = acc + bias[c];
        o[(size_t)r * H + c] = v;
        sum += v; ssq += v * v;
    }
    atomicAdd(&stats[c], sum);
    atomicAdd(&stats[H + c], ssq);
}

// BatchNorm (biased stats over B*N) + exact GELU; writes f32 in-place + bf16.
__global__ void k_bn_gelu(float* __restrict__ o, bf16* __restrict__ hbf,
                          const float* __restrict__ stats,
                          const float* __restrict__ gamma, const float* __restrict__ beta)
{
    int c = threadIdx.x;
    size_t idx = (size_t)blockIdx.x * H + c;
    const float cnt = (float)ROWS;
    float mu  = stats[c] / cnt;
    float var = stats[H + c] / cnt - mu * mu;
    float inv = rsqrtf(var + EPSV);
    float v = (o[idx] - mu) * inv * gamma[c] + beta[c];
    float g = 0.5f * v * (1.f + erff(v * 0.70710678118f));
    o[idx]   = g;
    hbf[idx] = (bf16)g;
}

// Graph readout: mean over nodes. grid = BATCH, block = H.
__global__ void k_readout(const float* __restrict__ o, float* __restrict__ temporal,
                          bf16* __restrict__ tempbf, int t)
{
    int b = blockIdx.x, c = threadIdx.x;
    float s = 0.f;
    for (int i = 0; i < NNODES; ++i)
        s += o[((size_t)b * NNODES + i) * H + c];
    s *= (1.f / NNODES);
    size_t m = (size_t)b * TSTEPS + t;
    temporal[m * H + c] = s;
    tempbf[m * H + c]   = (bf16)s;
}

// Attention: one block per (b, head, query). block = 64 (= T = hd).
__global__ __launch_bounds__(64) void k_attn(const float* __restrict__ qkv,
                                             bf16* __restrict__ aobf)
{
    int bid = blockIdx.x;           // ((b*NH + h)*T + q)
    int q   = bid & (TSTEPS - 1);
    int bh  = bid >> 6;
    int h   = bh & (NHEADS - 1);
    int b   = bh >> 2;
    int tid = threadIdx.x;

    __shared__ float qv[64], sc[64], red[64];
    const float* base = qkv + (size_t)(b * TSTEPS) * (3 * H);

    qv[tid] = base[(size_t)q * (3 * H) + h * 64 + tid];
    __syncthreads();

    const float* kr = base + (size_t)tid * (3 * H) + H + h * 64;
    float s = 0.f;
    for (int d = 0; d < 64; ++d) s += qv[d] * kr[d];
    s *= 0.125f;                    // 1/sqrt(64)
    sc[tid] = s; red[tid] = s;
    __syncthreads();
    for (int off = 32; off; off >>= 1) {
        if (tid < off) red[tid] = fmaxf(red[tid], red[tid + off]);
        __syncthreads();
    }
    float mx = red[0];
    __syncthreads();
    float e = expf(sc[tid] - mx);
    sc[tid] = e; red[tid] = e;
    __syncthreads();
    for (int off = 32; off; off >>= 1) {
        if (tid < off) red[tid] += red[tid + off];
        __syncthreads();
    }
    float inv = 1.f / red[0];
    __syncthreads();

    float acc = 0.f;
    for (int j = 0; j < 64; ++j)
        acc += sc[j] * base[(size_t)j * (3 * H) + 2 * H + h * 64 + tid];
    acc *= inv;
    aobf[((size_t)(b * TSTEPS) + q) * H + h * 64 + tid] = (bf16)acc;
}

// residual + LayerNorm. grid = MT, block = H.
__global__ __launch_bounds__(256) void k_add_ln(
    const float* __restrict__ temporal, const float* __restrict__ aop,
    const float* __restrict__ gamma, const float* __restrict__ beta,
    float* __restrict__ t2)
{
    int m = blockIdx.x, c = threadIdx.x;
    float v = temporal[(size_t)m * H + c] + aop[(size_t)m * H + c];
    __shared__ float sh[256];
    sh[c] = v; __syncthreads();
    for (int off = 128; off; off >>= 1) {
        if (c < off) sh[c] += sh[c + off];
        __syncthreads();
    }
    float mu = sh[0] * (1.f / H);
    __syncthreads();
    float d = v - mu;
    sh[c] = d * d; __syncthreads();
    for (int off = 128; off; off >>= 1) {
        if (c < off) sh[c] += sh[c + off];
        __syncthreads();
    }
    float var = sh[0] * (1.f / H);
    t2[(size_t)m * H + c] = d * rsqrtf(var + EPSV) * gamma[c] + beta[c];
}

// mean over T. grid = BATCH, block = H.
__global__ void k_pool(const float* __restrict__ t2, float* __restrict__ pooled)
{
    int b = blockIdx.x, c = threadIdx.x;
    float s = 0.f;
    for (int t = 0; t < TSTEPS; ++t)
        s += t2[((size_t)b * TSTEPS + t) * H + c];
    pooled[b * H + c] = s * (1.f / TSTEPS);
}

// h1 = gelu(pooled @ w_h1.T + b_h1). grid = BATCH, block = H.
__global__ void k_head1(const float* __restrict__ pooled, const float* __restrict__ w1,
                        const float* __restrict__ b1, float* __restrict__ h1)
{
    int b = blockIdx.x, c = threadIdx.x;
    float s = b1[c];
    const float* p = pooled + (size_t)b * H;
    const float* w = w1 + (size_t)c * H;
    for (int k = 0; k < H; ++k) s += p[k] * w[k];
    h1[b * H + c] = 0.5f * s * (1.f + erff(s * 0.70710678118f));
}

// out = h1 @ w_h2.T + b_h2
__global__ void k_head2(const float* __restrict__ h1, const float* __restrict__ w2,
                        const float* __restrict__ b2, float* __restrict__ out)
{
    int i = blockIdx.x * blockDim.x + threadIdx.x;
    if (i >= BATCH * NCLS) return;
    int b = i / NCLS, n = i % NCLS;
    float s = b2[n];
    for (int k = 0; k < H; ++k) s += h1[(size_t)b * H + k] * w2[(size_t)n * H + k];
    out[i] = s;
}

extern "C" void kernel_launch(void* const* d_in, const int* in_sizes, int n_in,
                              void* d_out, int out_size, void* d_ws, size_t ws_size,
                              hipStream_t stream)
{
    (void)in_sizes; (void)n_in; (void)out_size; (void)ws_size;

    const float* x       = (const float*)d_in[0];
    const float* adj     = (const float*)d_in[1];
    const float* w_in    = (const float*)d_in[2];
    const float* b_in    = (const float*)d_in[3];
    const float* gcn_w   = (const float*)d_in[4];
    const float* gcn_b   = (const float*)d_in[5];
    const float* gcn_g   = (const float*)d_in[6];
    const float* gcn_be  = (const float*)d_in[7];
    const float* w_qkv   = (const float*)d_in[8];
    const float* b_qkv   = (const float*)d_in[9];
    const float* w_o     = (const float*)d_in[10];
    const float* b_o     = (const float*)d_in[11];
    const float* ln_g    = (const float*)d_in[12];
    const float* ln_b    = (const float*)d_in[13];
    const float* w_h1    = (const float*)d_in[14];
    const float* b_h1    = (const float*)d_in[15];
    const float* w_h2    = (const float*)d_in[16];
    const float* b_h2    = (const float*)d_in[17];

    char* ws = (char*)d_ws;
    size_t p = 0;
    auto carve = [&](size_t bytes) -> char* {
        char* r = ws + p;
        p += (bytes + 255) & ~(size_t)255;
        return r;
    };

    bf16*  gcnw_bf  = (bf16*) carve((size_t)3 * H * H * sizeof(bf16));
    bf16*  wqkv_bf  = (bf16*) carve((size_t)3 * H * H * sizeof(bf16));
    bf16*  wo_bf    = (bf16*) carve((size_t)H * H * sizeof(bf16));
    bf16*  hbf      = (bf16*) carve((size_t)ROWS * H * sizeof(bf16));   // 8 MB
    float* sbuf     = (float*)carve((size_t)ROWS * H * sizeof(float));  // 16 MB
    float* obuf     = (float*)carve((size_t)ROWS * H * sizeof(float));  // 16 MB
    float* stats    = (float*)carve(2 * H * sizeof(float));
    float* temporal = (float*)carve((size_t)MT * H * sizeof(float));
    bf16*  tempbf   = (bf16*) carve((size_t)MT * H * sizeof(bf16));

    // buffer reuse after the timestep loop (hbf/sbuf/obuf are dead by then)
    float* qkvbuf = sbuf;                          // MT x 3H f32
    bf16*  aobf   = hbf;                           // MT x H bf16
    float* aop    = obuf;                          // MT x H f32
    float* t2     = obuf + (size_t)MT * H;         // MT x H f32
    float* pooled = obuf + (size_t)2 * MT * H;     // B x H
    float* h1     = pooled + (size_t)BATCH * H;    // B x H

    // --- convert weights to bf16 once ---
    k_f2bf<<<(3 * H * H + 255) / 256, 256, 0, stream>>>(gcn_w, gcnw_bf, 3 * H * H);
    k_f2bf<<<(3 * H * H + 255) / 256, 256, 0, stream>>>(w_qkv, wqkv_bf, 3 * H * H);
    k_f2bf<<<(H * H + 255) / 256, 256, 0, stream>>>(w_o, wo_bf, H * H);

    // --- per-timestep GCN stack ---
    for (int t = 0; t < TSTEPS; ++t) {
        k_in_proj<<<ROWS, H, 0, stream>>>(x, w_in, b_in, hbf, t);
        for (int l = 0; l < LCNT; ++l) {
            k_gemm_bf16<<<dim3(ROWS / 128, H / 64), 256, 0, stream>>>(
                hbf, gcnw_bf + (size_t)l * H * H, nullptr, sbuf, ROWS, H, H);
            k_zero<<<2, 256, 0, stream>>>(stats, 2 * H);
            k_aggregate<<<ROWS / 8, H, 0, stream>>>(
                sbuf, adj, gcn_b + (size_t)l * H, obuf, stats);
            k_bn_gelu<<<ROWS, H, 0, stream>>>(
                obuf, hbf, stats, gcn_g + (size_t)l * H, gcn_be + (size_t)l * H);
        }
        k_readout<<<BATCH, H, 0, stream>>>(obuf, temporal, tempbf, t);
    }

    // --- temporal attention ---
    k_gemm_bf16<<<dim3(MT / 128, (3 * H) / 64), 256, 0, stream>>>(
        tempbf, wqkv_bf, b_qkv, qkvbuf, MT, 3 * H, H);
    k_attn<<<BATCH * NHEADS * TSTEPS, 64, 0, stream>>>(qkvbuf, aobf);
    k_gemm_bf16<<<dim3(MT / 128, H / 64), 256, 0, stream>>>(
        aobf, wo_bf, b_o, aop, MT, H, H);
    k_add_ln<<<MT, H, 0, stream>>>(temporal, aop, ln_g, ln_b, t2);

    // --- head ---
    k_pool<<<BATCH, H, 0, stream>>>(t2, pooled);
    k_head1<<<BATCH, H, 0, stream>>>(pooled, w_h1, b_h1, h1);
    k_head2<<<(BATCH * NCLS + 63) / 64, 64, 0, stream>>>(h1, w_h2, b_h2, (float*)d_out);
}
